// MultiHeadAttention_54305566490675
// MI455X (gfx1250) — matrix operand order repaired and checked by reading, not compiled, not run on hardware
//
#include <hip/hip_runtime.h>
#include <hip/hip_bf16.h>
#include <math.h>

#define DEV __device__ __forceinline__

typedef __attribute__((ext_vector_type(16))) _Float16 v16h;
typedef __attribute__((ext_vector_type(8)))  float    v8f;
typedef __attribute__((ext_vector_type(4)))  float    v4f;
typedef __attribute__((ext_vector_type(4)))  unsigned u32x4;
typedef __attribute__((ext_vector_type(8)))  int      i32x8;
typedef __attribute__((ext_vector_type(4)))  int      i32x4;

constexpr int Bc = 4, Sc = 1024, Dc = 512, Hc = 8, DKc = 64;

DEV int laneid() { return threadIdx.x & 31; }

// LDS byte offset of a generic pointer into __shared__ (AS3 ptr value == offset)
typedef __attribute__((address_space(3))) const void lds_cv;
DEV unsigned lds_offset(const void* p) {
  return (unsigned)(unsigned long long)(lds_cv*)p;
}

// Per-lane async global->LDS copy of 16 bytes (CDNA5 ISA)
DEV void async_load_b128(unsigned lds_off, const void* gaddr) {
  asm volatile("global_load_async_to_lds_b128 %0, %1, off"
               :: "v"(lds_off), "v"(gaddr) : "memory");
}

// Stage one 16x64-f16 K tile (2 KB) into LDS: 4 async b128 ops per wave.
DEV void stage_k_tile(const _Float16* gsrc, unsigned lds_base) {
  int lane = laneid();
#pragma unroll
  for (int i = 0; i < 4; ++i) {
    int c   = lane + 32 * i;      // 16-byte chunk id 0..127
    int row = c >> 3;             // 0..15
    int blk = c & 7;              // 0..7 (8 halfs each)
    async_load_b128(lds_base + row * 128 + blk * 16,
                    (const void*)(gsrc + row * 64 + blk * 8));
  }
}

// TDM: DMA a 16-row x 64-half tile into LDS (CDNA5 ISA/8)
DEV void tdm_load_tile_16x64_f16(const _Float16* gsrc, unsigned lds_byte_off) {
#if __has_builtin(__builtin_amdgcn_tensor_load_to_lds)
  unsigned long long ga = (unsigned long long)(const void*)gsrc;
  u32x4 g0;
  g0[0] = 1u;                         // count=1, user mode, no gather
  g0[1] = lds_byte_off;               // lds_addr (bits 63:32)
  g0[2] = (unsigned)ga;               // global_addr lo
  g0[3] = (unsigned)((ga >> 32) & 0x01FFFFFFu) | (2u << 30);  // addr hi | type=2
  i32x8 g1;
  g1[0] = 0x00010000;                 // wg_mask=0 | data_size=1 (2B) | no flags
  g1[1] = (int)(64u << 16);           // tensor_dim0 = 64 (bits 79:48 lo16)
  g1[2] = (int)(16u << 16);           // tensor_dim1 = 16 (bits 111:80 lo16)
  g1[3] = (int)(64u << 16);           // tile_dim0 = 64 (bits 127:112)
  g1[4] = 16;                         // tile_dim1 = 16; tile_dim2 = 0
  g1[5] = 64;                         // tensor_dim0_stride = 64 (bits 207:160)
  g1[6] = 0;
  g1[7] = 0;
  i32x4 z4 = {0, 0, 0, 0};
  i32x8 z8 = {0, 0, 0, 0, 0, 0, 0, 0};
  // amdgpu-toolchain (clang-23) 6-arg form: (g0, g1, g2, g3, g4, cpol)
  __builtin_amdgcn_tensor_load_to_lds(g0, g1, z4, z4, z8, 0);
#if __has_builtin(__builtin_amdgcn_s_wait_tensorcnt)
  __builtin_amdgcn_s_wait_tensorcnt(0);
#else
  asm volatile("s_wait_tensorcnt 0x0" ::: "memory");
#endif
#else
  // Fallback: async-copy path
  stage_k_tile(gsrc, lds_byte_off);
  asm volatile("s_wait_asynccnt 0" ::: "memory");
#endif
}

// ---- WMMA fragment helpers (contiguous k order: element s of lane l holds k = 16*(l/16) + s) ----

// A fragment: 16x32 (MxK) f16.
template <typename T>
DEV v16h load_a_frag(const T* base, int ld) {
  int lane = laneid();
  int row  = lane & 15;
  int kh   = (lane >> 4) << 4;   // 0 or 16
  v16h a;
#pragma unroll
  for (int r = 0; r < 8; ++r) {
    int k = 2 * r + kh;
    a[2 * r]     = (_Float16)base[row * ld + k];
    a[2 * r + 1] = (_Float16)base[row * ld + k + 1];
  }
  return a;
}

// B fragment: 32x16 (KxN), source laid out [N][K]:  B[k][n] = base[n*ld + k]
template <typename T>
DEV v16h load_b_frag_t(const T* base, int ld) {
  int lane = laneid();
  int col  = lane & 15;
  int kh   = (lane >> 4) << 4;   // 0 or 16
  v16h b;
#pragma unroll
  for (int r = 0; r < 8; ++r) {
    int k = 2 * r + kh;
    b[2 * r]     = (_Float16)base[col * ld + k];
    b[2 * r + 1] = (_Float16)base[col * ld + k + 1];
  }
  return b;
}

// B fragment: 32x16 (KxN), source laid out [K][N]:  B[k][n] = base[k*ld + n]
template <typename T>
DEV v16h load_b_frag_n(const T* base, int ld) {
  int lane = laneid();
  int col  = lane & 15;
  int kh   = (lane >> 4) << 4;
  v16h b;
#pragma unroll
  for (int r = 0; r < 8; ++r) {
    int k = 2 * r + kh;
    b[2 * r]     = (_Float16)base[k * ld + col];
    b[2 * r + 1] = (_Float16)base[(k + 1) * ld + col];
  }
  return b;
}

DEV v8f wmma_f16(const v16h& a, const v16h& b, const v8f& c) {
  return __builtin_amdgcn_wmma_f32_16x16x32_f16(false, a, false, b, (short)0, c,
                                                false, false);
}

// ============================================================================
// Kernel 1: fused Q/K/V projections. Y = X @ W^T + bias  (K uses Wq: kq_same)
// ============================================================================
__global__ void proj_qkv_kernel(const float* __restrict__ q,
                                const float* __restrict__ k,
                                const float* __restrict__ v,
                                const float* __restrict__ Wq,
                                const float* __restrict__ Wv,
                                const float* __restrict__ bq,
                                const float* __restrict__ bv,
                                float* __restrict__ Yq,
                                float* __restrict__ Yk,
                                float* __restrict__ Yv) {
  int n0 = blockIdx.x * 16;
  int m0 = blockIdx.y * 16;
  int which = blockIdx.z;
  const float* X    = (which == 0) ? q : (which == 1) ? k : v;
  const float* W    = (which == 2) ? Wv : Wq;
  const float* bias = (which == 2) ? bv : bq;
  float* Y          = (which == 0) ? Yq : (which == 1) ? Yk : Yv;

  v8f c = {};
  for (int kt = 0; kt < Dc; kt += 32) {
    v16h a = load_a_frag(X + (size_t)m0 * Dc + kt, Dc);
    v16h b = load_b_frag_t(W + (size_t)n0 * Dc + kt, Dc);
    c = wmma_f16(a, b, c);
  }
  int lane = laneid();
  int col  = lane & 15;
  int rb   = (lane >> 4) << 3;
  float bb = bias[n0 + col];
#pragma unroll
  for (int r = 0; r < 8; ++r)
    Y[(size_t)(m0 + rb + r) * Dc + n0 + col] = c[r] + bb;
}

// ============================================================================
// Kernel 2: RoPE (interleaved pairs, base 10000) + reshape to [B,H,S,DK] f16.
// ============================================================================
__global__ void rope_reshape_kernel(const float* __restrict__ Yq,
                                    const float* __restrict__ Yk,
                                    const float* __restrict__ Yv,
                                    _Float16* __restrict__ qh,
                                    _Float16* __restrict__ kh,
                                    _Float16* __restrict__ vh) {
  int which = blockIdx.y;
  const float* Y = (which == 0) ? Yq : (which == 1) ? Yk : Yv;
  _Float16* O    = (which == 0) ? qh : (which == 1) ? kh : vh;

  int idx = blockIdx.x * blockDim.x + threadIdx.x;  // pair index
  int npairs = Bc * Sc * Dc / 2;
  if (idx >= npairs) return;

  int dp = idx % (Dc / 2);
  int bs = idx / (Dc / 2);
  int s  = bs % Sc;
  int b  = bs / Sc;
  int d0 = dp * 2;
  int h  = d0 / DKc;
  int dk = d0 % DKc;

  float x0 = Y[(size_t)bs * Dc + d0];
  float x1 = Y[(size_t)bs * Dc + d0 + 1];
  float y0 = x0, y1 = x1;
  if (which < 2) {
    int p = dk >> 1;
    float ang = (float)s * __powf(10000.0f, -2.0f * (float)p / (float)DKc);
    float sn, cs;
    __sincosf(ang, &sn, &cs);
    y0 = x0 * cs - x1 * sn;
    y1 = x1 * cs + x0 * sn;
  }
  size_t o = ((size_t)(b * Hc + h) * Sc + s) * DKc + dk;
  O[o]     = (_Float16)y0;
  O[o + 1] = (_Float16)y1;
}

// ============================================================================
// Kernel 3: attention core. One workgroup per (b, h, 16-row tile).
// 256 threads = 8 waves. LDS: 16x1024 f32 score tile (64KB) + per-wave
// double-buffered K staging (32KB, global_load_async_to_lds_b128) + TDM-staged
// Q tile (2KB, tensor_load_to_lds).
// ============================================================================
__global__ __launch_bounds__(256) void attn_kernel(
    const _Float16* __restrict__ qh, const _Float16* __restrict__ kh,
    const _Float16* __restrict__ vh, const float* __restrict__ gammas,
    float* __restrict__ attn, _Float16* __restrict__ outh) {
  __shared__ float sc[16][Sc];              // 64 KB score / attn tile
  __shared__ _Float16 kst[8][2][16][DKc];   // 32 KB K staging (wave x dbuf)
  __shared__ _Float16 qtile[16][DKc];       //  2 KB Q tile (TDM-loaded)

  int it = blockIdx.x;
  int h  = blockIdx.y;
  int b  = blockIdx.z;
  int i0 = it * 16;
  int wave = threadIdx.x >> 5;
  int lane = threadIdx.x & 31;
  int col  = lane & 15;
  int rb   = (lane >> 4) << 3;

  const _Float16* qhd = qh + ((size_t)(b * Hc + h) * Sc) * DKc;
  const _Float16* khd = kh + ((size_t)(b * Hc + h) * Sc) * DKc;
  const _Float16* vhd = vh + ((size_t)(b * Hc + h) * Sc) * DKc;

  // --- Phase 0: TDM DMA of the Q row tile into LDS (one wave issues) ---
  if (wave == 0)
    tdm_load_tile_16x64_f16(qhd + (size_t)i0 * DKc, lds_offset(&qtile[0][0]));
  __syncthreads();

  // --- Phase 1: score tile via async-staged K + WMMA, double buffered ---
  v16h aq0 = load_a_frag(&qtile[0][0], DKc);
  v16h aq1 = load_a_frag(&qtile[0][0] + 32, DKc);

  unsigned kbase = lds_offset(&kst[wave][0][0][0]);
  stage_k_tile(khd + (size_t)(wave * 16) * DKc, kbase);   // tile t=0 -> buf 0
  int buf = 0;
  for (int t = 0; t < 8; ++t) {
    int j0 = (wave + t * 8) * 16;
    if (t < 7) {
      int j0n = (wave + (t + 1) * 8) * 16;
      stage_k_tile(khd + (size_t)j0n * DKc, kbase + (buf ^ 1) * 2048);
      asm volatile("s_wait_asynccnt 4" ::: "memory");  // tile t complete
    } else {
      asm volatile("s_wait_asynccnt 0" ::: "memory");
    }
    const _Float16* ktile = &kst[wave][buf][0][0];
    v16h b0 = load_b_frag_t(ktile, DKc);
    v16h b1 = load_b_frag_t(ktile + 32, DKc);
    v8f c = {};
    c = wmma_f16(aq0, b0, c);
    c = wmma_f16(aq1, b1, c);
#pragma unroll
    for (int r = 0; r < 8; ++r)
      sc[rb + r][j0 + col] = c[r] * 0.125f;  // 1/sqrt(64)
    buf ^= 1;
  }
  __syncthreads();

  // --- Phase 2: per-row gating + softmax (2 rows per wave) ---
  float gamma = gammas[h];
  float g = -log1pf(__expf(gamma));        // -softplus(gamma)
  float slope = exp2f(-(float)(h + 1));    // ALiBi slopes for H=8

  for (int rr = 0; rr < 2; ++rr) {
    int r = wave + rr * 8;
    int i = i0 + r;

    // softmax 1: max
    float m = -3.0e38f;
    for (int ck = 0; ck < Sc / 32; ++ck) {
      int j = ck * 32 + lane;
      float s = (j <= i) ? sc[r][j] : -3.0e38f;
      m = fmaxf(m, s);
    }
    for (int off = 16; off > 0; off >>= 1) m = fmaxf(m, __shfl_xor(m, off));
    // softmax 1: sum
    float sum = 0.f;
    for (int ck = 0; ck < Sc / 32; ++ck) {
      int j = ck * 32 + lane;
      if (j <= i) sum += __expf(sc[r][j] - m);
    }
    for (int off = 16; off > 0; off >>= 1) sum += __shfl_xor(sum, off);
    float inv = 1.0f / sum;

    // cumsum of s_ -> distance-decay gate -> ALiBi
    float carry = 0.f;
    for (int ck = 0; ck < Sc / 32; ++ck) {
      int j = ck * 32 + lane;
      float p = (j <= i) ? __expf(sc[r][j] - m) * inv : 0.f;
      float x = p;
      for (int off = 1; off < 32; off <<= 1) {  // wave32 inclusive scan
        float y = __shfl_up(x, off);
        if (lane >= off) x += y;
      }
      float distcum = carry + x;               // inclusive cumsum of s_
      float pos  = fabsf((float)j - (float)i);
      float dist = sqrtf(fmaxf((1.0f - distcum) * pos, 0.f));  // disttot==1
      float te   = __expf(dist * g);
      te = fminf(fmaxf(te, 1e-5f), 1e5f);
      sc[r][j] = sc[r][j] * te + slope * (float)j;
      carry += __shfl(x, 31);
    }

    // softmax 2
    float m2 = -3.0e38f;
    for (int ck = 0; ck < Sc / 32; ++ck) {
      int j = ck * 32 + lane;
      float s = (j <= i) ? sc[r][j] : -3.0e38f;
      m2 = fmaxf(m2, s);
    }
    for (int off = 16; off > 0; off >>= 1) m2 = fmaxf(m2, __shfl_xor(m2, off));
    float sum2 = 0.f;
    for (int ck = 0; ck < Sc / 32; ++ck) {
      int j = ck * 32 + lane;
      if (j <= i) sum2 += __expf(sc[r][j] - m2);
    }
    for (int off = 16; off > 0; off >>= 1) sum2 += __shfl_xor(sum2, off);
    float inv2 = 1.0f / sum2;

    for (int ck = 0; ck < Sc / 32; ++ck) {
      int j = ck * 32 + lane;
      float a_ = (j <= i) ? __expf(sc[r][j] - m2) * inv2 : 0.f;
      sc[r][j] = a_;   // keep in LDS for NT writeback + attn@V
    }

    // dominant HBM stream: full attn row, b128 non-temporal stores
    float* arow = attn + ((size_t)(b * Hc + h) * Sc + i) * Sc;
    for (int ck = 0; ck < Sc / 128; ++ck) {
      int j = ck * 128 + lane * 4;
      v4f vals = *(const v4f*)&sc[r][j];
      __builtin_nontemporal_store(vals, (v4f*)(arow + j));
    }
  }
  __syncthreads();

  // --- Phase 3: out = attn @ V (waves 0..3, one 16-wide N tile each) ---
  if (wave < 4) {
    int n0 = wave * 16;
    v8f c = {};
    for (int kt = 0; kt < Sc; kt += 32) {
      __builtin_prefetch(vhd + (size_t)(kt + 64) * DKc + n0, 0, 0);
      v16h a  = load_a_frag(&sc[0][kt], Sc);                    // from LDS
      v16h b2 = load_b_frag_n(vhd + (size_t)kt * DKc + n0, DKc);
      c = wmma_f16(a, b2, c);
    }
    _Float16* orow = outh + ((size_t)b * Sc + i0) * Dc + h * DKc + n0;
#pragma unroll
    for (int r = 0; r < 8; ++r)
      orow[(size_t)(rb + r) * Dc + col] = (_Float16)c[r];
  }
}

// ============================================================================
// Kernel 4: output projection  out = outh @ Wo^T + bo
// ============================================================================
__global__ void outproj_kernel(const _Float16* __restrict__ outh,
                               const float* __restrict__ Wo,
                               const float* __restrict__ bo,
                               float* __restrict__ out) {
  int n0 = blockIdx.x * 16;
  int m0 = blockIdx.y * 16;
  v8f c = {};
  for (int kt = 0; kt < Dc; kt += 32) {
    v16h a = load_a_frag(outh + (size_t)m0 * Dc + kt, Dc);
    v16h b = load_b_frag_t(Wo + (size_t)n0 * Dc + kt, Dc);
    c = wmma_f16(a, b, c);
  }
  int lane = laneid();
  int col  = lane & 15;
  int rb   = (lane >> 4) << 3;
  float bb = bo[n0 + col];
#pragma unroll
  for (int r = 0; r < 8; ++r)
    out[(size_t)(m0 + rb + r) * Dc + n0 + col] = c[r] + bb;
}

// ============================================================================
extern "C" void kernel_launch(void* const* d_in, const int* in_sizes, int n_in,
                              void* d_out, int out_size, void* d_ws,
                              size_t ws_size, hipStream_t stream) {
  const float* q      = (const float*)d_in[0];
  const float* k      = (const float*)d_in[1];
  const float* v      = (const float*)d_in[2];
  /* d_in[3] = mask (causal tril; handled analytically) */
  const float* Wq     = (const float*)d_in[4];
  const float* bq     = (const float*)d_in[5];
  const float* Wv     = (const float*)d_in[6];
  const float* bv     = (const float*)d_in[7];
  const float* Wo     = (const float*)d_in[8];
  const float* bo     = (const float*)d_in[9];
  const float* gammas = (const float*)d_in[10];

  char* ws = (char*)d_ws;
  size_t szY = (size_t)Bc * Sc * Dc * sizeof(float);          // 8 MB each
  size_t szH = (size_t)Bc * Hc * Sc * DKc * sizeof(_Float16); // 4 MB each
  float*    Yq   = (float*)(ws);
  float*    Yk   = (float*)(ws + szY);
  float*    Yv   = (float*)(ws + 2 * szY);
  _Float16* qh   = (_Float16*)(ws + 3 * szY);
  _Float16* kh   = (_Float16*)(ws + 3 * szY + szH);
  _Float16* vh   = (_Float16*)(ws + 3 * szY + 2 * szH);
  _Float16* outh = (_Float16*)(ws + 3 * szY + 3 * szH);

  float* out  = (float*)d_out;
  float* attn = out + (size_t)Bc * Sc * Dc;  // tuple: (out, attn) concatenated

  dim3 g1(Dc / 16, (Bc * Sc) / 16, 3);
  proj_qkv_kernel<<<g1, 32, 0, stream>>>(q, k, v, Wq, Wv, bq, bv, Yq, Yk, Yv);

  int npairs = Bc * Sc * Dc / 2;
  dim3 g2((npairs + 255) / 256, 3, 1);
  rope_reshape_kernel<<<g2, 256, 0, stream>>>(Yq, Yk, Yv, qh, kh, vh);

  dim3 g3(Sc / 16, Hc, Bc);
  attn_kernel<<<g3, 256, 0, stream>>>(qh, kh, vh, gammas, attn, outh);

  dim3 g4(Dc / 16, (Bc * Sc) / 16, 1);
  outproj_kernel<<<g4, 32, 0, stream>>>(outh, Wo, bo, out);
}
